// GCN_20091857011065
// MI455X (gfx1250) — compile-verified, matrix operand order checked by reading, not run on hardware
//
#include <hip/hip_runtime.h>
#include <hip/hip_bf16.h>

typedef __attribute__((ext_vector_type(16))) __bf16 v16bf;
typedef __attribute__((ext_vector_type(8)))  float  v8f;

#define FEAT 128
#define CEIL(a, b) (((a) + (b) - 1) / (b))

// ---------------------------------------------------------------- utilities
__global__ __launch_bounds__(256) void zero_kernel(float* __restrict__ p, int n) {
  int i = blockIdx.x * blockDim.x + threadIdx.x;
  if (i < n) p[i] = 0.0f;
}

__global__ __launch_bounds__(256) void deg_kernel(const int* __restrict__ dst,
                                                  float* __restrict__ deg, int e) {
  int i = blockIdx.x * blockDim.x + threadIdx.x;
  if (i < e) atomicAdd(&deg[dst[i]], 1.0f);
}

// deg -> dinv = rsqrt(deg + 1)   (+1 = self loop; always > 0)
__global__ __launch_bounds__(256) void dinv_kernel(float* __restrict__ d, int n) {
  int i = blockIdx.x * blockDim.x + threadIdx.x;
  if (i < n) d[i] = rsqrtf(d[i] + 1.0f);
}

// ------------------------------------------------- GEMM: Y[n,128] = X[n,128] @ W[128,128]
// One wave computes one 16x16 output tile via 4 chained v_wmma_f32_16x16x32_bf16.
// Block = 8 waves = 16 rows x 128 cols. fp32 inputs cast to bf16, fp32 accumulate.
//
// A frag (16-bit 16x32, ISA 7.12.2): lane l holds row m=l&15.
//   lanes 0-15 : VGPR0-3 -> K=0..7 (2/VGPR), VGPR4-7 -> K=16..23
//   lanes 16-31: K=8..15 and K=24..31  (kb offset = 8)
// B frag mirrored: lane l holds column n=l&15, same K pattern.
// D: lane l holds column n=l&15; VGPR r -> row r + 8*(l>>4).
__global__ __launch_bounds__(256) void gemm_wmma_kernel(const float* __restrict__ X,
                                                        const float* __restrict__ W,
                                                        float* __restrict__ Y, int nrows) {
  __builtin_prefetch(W, 0, 3);  // -> global_prefetch_b8 (W = 64KB, pin in cache)
  const int wave = threadIdx.x >> 5;
  const int lane = threadIdx.x & 31;
  const int row0 = blockIdx.x << 4;
  const int m    = lane & 15;
  const int kh   = lane >> 4;        // half-selector: K base offset 0 or 8
  const int col0 = wave << 4;
  const int nc   = col0 + m;         // B/D column owned by this lane

  int row = row0 + m;
  if (row >= nrows) row = nrows - 1;           // clamp (N % 16 == 0 anyway)
  const float* __restrict__ xrow = X + (size_t)row * FEAT;

  v8f acc = {};
#pragma unroll
  for (int k0 = 0; k0 < FEAT; k0 += 32) {
    const int kb = k0 + (kh << 3);
    v16bf a, b;
#pragma unroll
    for (int e = 0; e < 8; ++e) {
      a[e]     = (__bf16)xrow[kb + e];
      a[e + 8] = (__bf16)xrow[kb + 16 + e];
      b[e]     = (__bf16)W[(kb + e) * FEAT + nc];
      b[e + 8] = (__bf16)W[(kb + 16 + e) * FEAT + nc];
    }
    acc = __builtin_amdgcn_wmma_f32_16x16x32_bf16(
        /*neg_a=*/false, a, /*neg_b=*/false, b,
        /*c_mod=*/(short)0, acc, /*reuse_a=*/false, /*reuse_b=*/false);
  }
#pragma unroll
  for (int r = 0; r < 8; ++r) {
    int rr = row0 + (kh << 3) + r;
    if (rr < nrows) Y[(size_t)rr * FEAT + nc] = acc[r];
  }
}

// ---------------- AGG[i,:] = bias + H[i,:] * dinv[i]^2   (bias + self-loop term)
__global__ __launch_bounds__(256) void agg_init_kernel(const float* __restrict__ H,
                                                       const float* __restrict__ dinv,
                                                       const float* __restrict__ bias,
                                                       float* __restrict__ AGG, int n) {
  int t = blockIdx.x * blockDim.x + threadIdx.x;  // over n*32 float4s
  if (t >= n * 32) return;
  int node = t >> 5;
  int q = t & 31;
  float dn = dinv[node];
  dn *= dn;
  float4 h  = ((const float4*)H)[t];
  float4 b4 = ((const float4*)bias)[q];
  float4 r;
  r.x = b4.x + h.x * dn;
  r.y = b4.y + h.y * dn;
  r.z = b4.z + h.z * dn;
  r.w = b4.w + h.w * dn;
  ((float4*)AGG)[t] = r;
}

// ---------------- edge scatter: AGG[dst,:] += H[src,:] * dinv[src]*dinv[dst]
// one wave (32 lanes) per edge, float4 per lane, f32 global atomics.
__global__ __launch_bounds__(256) void scatter_kernel(const int* __restrict__ src,
                                                      const int* __restrict__ dst,
                                                      const float* __restrict__ dinv,
                                                      const float* __restrict__ H,
                                                      float* __restrict__ AGG, int e) {
  int edge = (blockIdx.x * blockDim.x + threadIdx.x) >> 5;
  int lane = threadIdx.x & 31;
  if (edge >= e) return;
  int s = src[edge], d = dst[edge];
  float nrm = dinv[s] * dinv[d];
  float4 v = ((const float4*)(H + (size_t)s * FEAT))[lane];
  float* ad = AGG + (size_t)d * FEAT + lane * 4;
  atomicAdd(ad + 0, v.x * nrm);
  atomicAdd(ad + 1, v.y * nrm);
  atomicAdd(ad + 2, v.z * nrm);
  atomicAdd(ad + 3, v.w * nrm);
}

__global__ __launch_bounds__(256) void relu_kernel(float* __restrict__ p, int n4) {
  int t = blockIdx.x * blockDim.x + threadIdx.x;
  if (t >= n4) return;
  float4 v = ((float4*)p)[t];
  v.x = fmaxf(v.x, 0.0f);
  v.y = fmaxf(v.y, 0.0f);
  v.z = fmaxf(v.z, 0.0f);
  v.w = fmaxf(v.w, 0.0f);
  ((float4*)p)[t] = v;
}

// ---------------- segment-sum pooling: one wave per node
__global__ __launch_bounds__(256) void pool_kernel(const float* __restrict__ H,
                                                   const int* __restrict__ batch,
                                                   float* __restrict__ sums,
                                                   float* __restrict__ counts, int n) {
  int node = (blockIdx.x * blockDim.x + threadIdx.x) >> 5;
  int lane = threadIdx.x & 31;
  if (node >= n) return;
  int g = batch[node];
  if (lane == 0) atomicAdd(&counts[g], 1.0f);
  float4 v = ((const float4*)(H + (size_t)node * FEAT))[lane];
  float* sg = sums + (size_t)g * FEAT + lane * 4;
  atomicAdd(sg + 0, v.x);
  atomicAdd(sg + 1, v.y);
  atomicAdd(sg + 2, v.z);
  atomicAdd(sg + 3, v.w);
}

// ---------------- head: out[g, j] = (sums[g,:] / max(cnt,1)) . Wl[:, j] + bl[j]
__global__ __launch_bounds__(256) void head_kernel(const float* __restrict__ sums,
                                                   const float* __restrict__ counts,
                                                   const float* __restrict__ Wl,
                                                   const float* __restrict__ bl,
                                                   float* __restrict__ out, int ngj) {
  int idx = blockIdx.x * blockDim.x + threadIdx.x;
  if (idx >= ngj) return;
  int g = idx >> 1, j = idx & 1;
  float cnt = counts[g];
  float inv = 1.0f / fmaxf(cnt, 1.0f);
  const float* sg = sums + (size_t)g * FEAT;
  float acc = 0.0f;
#pragma unroll 4
  for (int c = 0; c < FEAT; ++c) acc += sg[c] * Wl[c * 2 + j];
  out[idx] = acc * inv + bl[j];
}

// ---------------------------------------------------------------- driver
extern "C" void kernel_launch(void* const* d_in, const int* in_sizes, int n_in,
                              void* d_out, int out_size, void* d_ws, size_t ws_size,
                              hipStream_t stream) {
  const float* x   = (const float*)d_in[0];
  const int*   ei  = (const int*)d_in[1];   // [2, E] flat
  const int*   bat = (const int*)d_in[2];
  const float* W1  = (const float*)d_in[3];
  const float* b1  = (const float*)d_in[4];
  const float* W2  = (const float*)d_in[5];
  const float* b2  = (const float*)d_in[6];
  const float* Wl  = (const float*)d_in[7];
  const float* bl  = (const float*)d_in[8];
  float* out = (float*)d_out;

  const int N = in_sizes[0] / FEAT;   // 100000
  const int E = in_sizes[1] / 2;      // 1600000
  const int G = out_size / 2;         // 1024

  const int* src = ei;
  const int* dst = ei + E;

  // workspace layout
  char* ws = (char*)d_ws;
  float* dinv   = (float*)(ws);                                   // N
  float* bufH   = (float*)(ws + 524288);                          // N*128
  float* bufA   = (float*)(ws + 524288 + (size_t)N * FEAT * 4);   // N*128
  float* sums   = (float*)((char*)bufA + (size_t)N * FEAT * 4);   // G*128
  float* counts = (float*)((char*)sums + (size_t)G * FEAT * 4);   // G

  const int nq = N * 32;  // float4 count per N*128 buffer

  // degrees -> dinv
  zero_kernel<<<CEIL(N, 256), 256, 0, stream>>>(dinv, N);
  deg_kernel<<<CEIL(E, 256), 256, 0, stream>>>(dst, dinv, E);
  dinv_kernel<<<CEIL(N, 256), 256, 0, stream>>>(dinv, N);

  // ---- layer 1
  gemm_wmma_kernel<<<CEIL(N, 16), 256, 0, stream>>>(x, W1, bufH, N);
  agg_init_kernel<<<CEIL(nq, 256), 256, 0, stream>>>(bufH, dinv, b1, bufA, N);
  scatter_kernel<<<CEIL(E * 32, 256), 256, 0, stream>>>(src, dst, dinv, bufH, bufA, E);
  relu_kernel<<<CEIL(nq, 256), 256, 0, stream>>>(bufA, nq);

  // ---- layer 2
  gemm_wmma_kernel<<<CEIL(N, 16), 256, 0, stream>>>(bufA, W2, bufH, N);
  agg_init_kernel<<<CEIL(nq, 256), 256, 0, stream>>>(bufH, dinv, b2, bufA, N);
  scatter_kernel<<<CEIL(E * 32, 256), 256, 0, stream>>>(src, dst, dinv, bufH, bufA, E);
  relu_kernel<<<CEIL(nq, 256), 256, 0, stream>>>(bufA, nq);

  // ---- mean pool + linear head
  zero_kernel<<<CEIL(G * FEAT, 256), 256, 0, stream>>>(sums, G * FEAT);
  zero_kernel<<<CEIL(G, 256), 256, 0, stream>>>(counts, G);
  pool_kernel<<<CEIL(N * 32, 256), 256, 0, stream>>>(bufA, bat, sums, counts, N);
  head_kernel<<<CEIL(G * 2, 256), 256, 0, stream>>>(sums, counts, Wl, bl, out, G * 2);
}
